// LanguageAttention_55929064129330
// MI455X (gfx1250) — matrix operand order repaired and checked by reading, not compiled, not run on hardware
//
#include <hip/hip_runtime.h>

// ---------------------------------------------------------------------------
// LanguageAttention for MI455X (gfx1250): bf16 WMMA GEMMs + flash attention,
// with Tensor Data Mover (TDM) async tile staging into LDS (double-buffered).
// Shapes fixed by the reference: B=2, S=2048, H=2048, NH=16, NKV=4, HD=128.
// ---------------------------------------------------------------------------

typedef unsigned short u16;
typedef unsigned int   u32;
typedef unsigned long long u64;
typedef __attribute__((ext_vector_type(16))) __bf16 v16bf;
typedef __attribute__((ext_vector_type(8)))  float  v8f;
typedef __attribute__((ext_vector_type(4)))  int    i32x4;
typedef __attribute__((ext_vector_type(8)))  int    i32x8;

union Frag16 { uint4 u[2]; v16bf v; };          // 32B: one WMMA A or B fragment
union FragC  { v8f v; float f[8]; };            // 32B: one WMMA C/D fragment

#define B_     2
#define S_     2048
#define H_     2048
#define NH_    16
#define NKV_   4
#define HD_    128
#define QKN_   (NH_ * HD_)    // 2048
#define KVN_   (NKV_ * HD_)   // 512
#define BS_    (B_ * S_)      // 4096

__device__ __forceinline__ u16 f2bf(float x) {
    u32 u = __builtin_bit_cast(u32, x);
    u += 0x7fffu + ((u >> 16) & 1u);            // round-to-nearest-even
    return (u16)(u >> 16);
}
__device__ __forceinline__ float bf2f(u16 h) {
    u32 u = ((u32)h) << 16;
    return __builtin_bit_cast(float, u);
}

// ---------------------------------------------------------------------------
// TDM: async DMA of a 2D bf16 tile (d1 rows x d0 elems, row stride `stride`
// elems) from global memory into LDS at byte offset lds_off. Issued per-wave
// (EXEC ignored) -> caller must gate to one wave per workgroup. Tracked by
// TENSORcnt. Descriptor per CDNA5 ISA ch.8: group0 = count/lds/global/type,
// group1 = data_size, tensor dims (== tile dims, always in-bounds), tile
// dims, dim0 stride. Groups 2/3 zero (2D tensor).
// ---------------------------------------------------------------------------
__device__ __forceinline__ void tdm_load_2d(u32 lds_off, const void* g,
                                            u32 d0, u32 d1, u32 stride) {
    u64 ga = (u64)g;
    i32x4 g0;
    g0[0] = 1;                                    // count=1, user descriptor
    g0[1] = (int)lds_off;                         // lds_addr (bytes)
    g0[2] = (int)(u32)ga;                         // global_addr[31:0]
    g0[3] = (int)(((u32)(ga >> 32) & 0x01FFFFFFu) | (2u << 30)); // addr[56:32]|type=2
    i32x8 g1;
    g1[0] = 0x00010000;                           // wg_mask=0, data_size=1 (2B)
    g1[1] = (int)(d0 << 16);                      // tensor_dim0[79:48] lo
    g1[2] = (int)((d0 >> 16) | (d1 << 16));       // tensor_dim0 hi | tensor_dim1 lo
    g1[3] = (int)((d1 >> 16) | (d0 << 16));       // tensor_dim1 hi | tile_dim0
    g1[4] = (int)d1;                              // tile_dim1 (tile_dim2=0)
    g1[5] = (int)stride;                          // tensor_dim0_stride[191:160]
    g1[6] = 0;                                    // stride hi16, dim1_stride lo
    g1[7] = 0;
    i32x4 z; z[0] = 0; z[1] = 0; z[2] = 0; z[3] = 0;
    asm volatile("tensor_load_to_lds %0, %1, %2, %3"
                 :: "s"(g0), "s"(g1), "s"(z), "s"(z)
                 : "memory");
}

__device__ __forceinline__ u32 lds_off_of(const void* p) {
    return (u32)(u64)p;                           // flat LDS addr: offset in [31:0]
}

// ---------------------------------------------------------------------------
// fp32 -> bf16, straight copy
// ---------------------------------------------------------------------------
__global__ __launch_bounds__(256) void cvt_bf16(const float* __restrict__ in,
                                                u16* __restrict__ out, int n) {
    int i = blockIdx.x * 256 + threadIdx.x;
    if (i < n) out[i] = f2bf(in[i]);
}

// ---------------------------------------------------------------------------
// fp32 [K][N] -> bf16 [N][K] (transpose so GEMM B-fragments are contiguous)
// ---------------------------------------------------------------------------
__global__ __launch_bounds__(256) void cvt_bf16_t(const float* __restrict__ in,
                                                  u16* __restrict__ out,
                                                  int K, int N) {
    int i = blockIdx.x * 256 + threadIdx.x;
    if (i < K * N) {
        int k = i / N, n = i - k * N;
        out[(size_t)n * K + k] = f2bf(in[i]);
    }
}

// ---------------------------------------------------------------------------
// bf16 GEMM: C[M][N] = A[M][K] * Bt[N][K]^T  via v_wmma_f32_16x16x32_bf16
// Block tile 128x128x64, 256 threads = 8 waves, wave tile 32x64.
// K-slabs staged into LDS by TDM (wave 0), double-buffered.
// ---------------------------------------------------------------------------
template <int OUTF32>
__global__ __launch_bounds__(256) void gemm_bf16_wmma(
    const u16* __restrict__ A, const u16* __restrict__ Bt,
    void* __restrict__ C, int M, int N, int K) {
    __shared__ __align__(16) u16 As[2][128 * 64];
    __shared__ __align__(16) u16 Bs[2][128 * 64];

    const int tid  = threadIdx.x;
    const int lane = tid & 31;
    const int wid  = tid >> 5;
    const int lh   = lane & 15;          // column/row within 16-wide subtile
    const int hi   = lane >> 4;          // lane half selects K sub-ranges
    const int bm   = blockIdx.y * 128;
    const int bn   = blockIdx.x * 128;
    const int wm   = (wid & 3) * 32;     // 4 waves along M
    const int wn   = (wid >> 2) * 64;    // 2 waves along N

    FragC acc[2][4];
#pragma unroll
    for (int m = 0; m < 2; ++m)
#pragma unroll
        for (int n = 0; n < 4; ++n)
#pragma unroll
            for (int r = 0; r < 8; ++r) acc[m][n].f[r] = 0.0f;

    const int nkb = K >> 6;
    if (wid == 0) {                      // prologue: fill buffer 0
        tdm_load_2d(lds_off_of(&As[0][0]), A + (size_t)bm * K, 64, 128, K);
        tdm_load_2d(lds_off_of(&Bs[0][0]), Bt + (size_t)bn * K, 64, 128, K);
    }

    for (int kb = 0; kb < nkb; ++kb) {
        const int cur = kb & 1;
        if (wid == 0) {
            if (kb + 1 < nkb) {          // issue next slab, wait current done
                tdm_load_2d(lds_off_of(&As[cur ^ 1][0]),
                            A + (size_t)bm * K + (kb + 1) * 64, 64, 128, K);
                tdm_load_2d(lds_off_of(&Bs[cur ^ 1][0]),
                            Bt + (size_t)bn * K + (kb + 1) * 64, 64, 128, K);
                __builtin_amdgcn_s_wait_tensorcnt(2);
            } else {
                __builtin_amdgcn_s_wait_tensorcnt(0);
            }
        }
        __syncthreads();

        const u16* Asb = &As[cur][0];
        const u16* Bsb = &Bs[cur][0];
#pragma unroll
        for (int ks = 0; ks < 2; ++ks) {
            // preload all fragments, then a back-to-back wmma chain
            Frag16 af[2], bfr[4];
#pragma unroll
            for (int m = 0; m < 2; ++m) {
                int row = wm + m * 16 + lh;
                int k0  = ks * 32 + (hi ? 8 : 0);  // A layout K sub-ranges
                af[m].u[0] = *(const uint4*)(Asb + row * 64 + k0);
                af[m].u[1] = *(const uint4*)(Asb + row * 64 + k0 + 16);
            }
#pragma unroll
            for (int n = 0; n < 4; ++n) {
                int col = wn + n * 16 + lh;
                int k0  = ks * 32 + (hi ? 8 : 0);  // B layout K sub-ranges
                bfr[n].u[0] = *(const uint4*)(Bsb + col * 64 + k0);
                bfr[n].u[1] = *(const uint4*)(Bsb + col * 64 + k0 + 16);
            }
#pragma unroll
            for (int n = 0; n < 4; ++n)
#pragma unroll
                for (int m = 0; m < 2; ++m)
                    acc[m][n].v = __builtin_amdgcn_wmma_f32_16x16x32_bf16(
                        false, af[m].v, false, bfr[n].v, (short)0, acc[m][n].v,
                        false, false);
        }
        __syncthreads();
    }

#pragma unroll
    for (int m = 0; m < 2; ++m)
#pragma unroll
        for (int n = 0; n < 4; ++n)
#pragma unroll
            for (int r = 0; r < 8; ++r) {
                int row = bm + wm + m * 16 + r + hi * 8;  // C layout
                int col = bn + wn + n * 16 + lh;
                float v = acc[m][n].f[r];
                if (OUTF32)
                    ((float*)C)[(size_t)row * N + col] = v;
                else
                    ((u16*)C)[(size_t)row * N + col] = f2bf(v);
            }
}

// ---------------------------------------------------------------------------
// RoPE + reshape:
//   qproj[BS][2048] -> Qb[(b*NH+h)*S + s][HD]      (row-major, WMMA A for QK^T)
//   kproj[BS][512]  -> Kb[(b*NKV+g)*S + s][HD]     (row-major, WMMA B for QK^T)
//   vproj[BS][512]  -> Vt[(b*NKV+g)*HD + d][S]     (transposed, WMMA B for P*V)
// ---------------------------------------------------------------------------
__global__ __launch_bounds__(128) void rope_reshape(
    const u16* __restrict__ qproj, const u16* __restrict__ kproj,
    const u16* __restrict__ vproj, u16* __restrict__ Qb,
    u16* __restrict__ Kb, u16* __restrict__ Vt) {
    const int bs = blockIdx.x;
    const int d  = threadIdx.x;
    const int b  = bs >> 11;
    const int s  = bs & (S_ - 1);
    const int d2 = d & 63;
    // inv_freq = 10000^(-d2/64) = exp2(-d2 * log2(10000)/64)
    const float theta = (float)s * exp2f(-(float)d2 * 0.20762050593046083f);
    const float c  = cosf(theta);
    const float sn = sinf(theta);
    const int   partner = (d < 64) ? d + 64 : d - 64;
    const float sgn     = (d < 64) ? -1.0f : 1.0f;

#pragma unroll
    for (int h = 0; h < NH_; ++h) {
        const u16* qp = qproj + (size_t)bs * QKN_ + h * HD_;
        float x = bf2f(qp[d]), xp = bf2f(qp[partner]);
        Qb[((size_t)(b * NH_ + h) * S_ + s) * HD_ + d] = f2bf(x * c + sgn * xp * sn);
    }
#pragma unroll
    for (int g = 0; g < NKV_; ++g) {
        const u16* kp = kproj + (size_t)bs * KVN_ + g * HD_;
        float x = bf2f(kp[d]), xp = bf2f(kp[partner]);
        Kb[((size_t)(b * NKV_ + g) * S_ + s) * HD_ + d] = f2bf(x * c + sgn * xp * sn);
        const u16* vp = vproj + (size_t)bs * KVN_ + g * HD_;
        Vt[((size_t)(b * NKV_ + g) * HD_ + d) * S_ + s] = vp[d];
    }
}

// ---------------------------------------------------------------------------
// Flash attention (causal, GQA). 128 threads = 4 waves; block owns 128 query
// rows of one (b, h); each wave owns 32 rows. K/V tiles (32 keys) staged into
// LDS by TDM (wave 0), double-buffered across the key loop. Online softmax in
// the WMMA C-layout (lanes = columns, VGPR idx = row); row reductions are
// 16-lane shfl_xor butterflies.
// ---------------------------------------------------------------------------
__global__ __launch_bounds__(128) void flash_attn(
    const u16* __restrict__ Qb, const u16* __restrict__ Kb,
    const u16* __restrict__ Vt, u16* __restrict__ attn) {
    __shared__ __align__(16) u16 Ks[2][32 * 128];   // [krow][d]
    __shared__ __align__(16) u16 Vts[2][128 * 32];  // [d][krow]
    __shared__ __align__(16) u16 Ps[4 * 32 * 32];   // per-wave P scratch

    const int tid  = threadIdx.x;
    const int lane = tid & 31;
    const int wave = tid >> 5;
    const int lh   = lane & 15;
    const int hi   = lane >> 4;
    const int rofs = hi * 8;                        // C-layout row offset

    const int qblk = blockIdx.x * 128;
    const int h    = blockIdx.y;
    const int b    = blockIdx.z;
    const int kv   = h >> 2;                        // GROUPS = NH/NKV = 4
    const int waveQ = qblk + wave * 32;

    const u16* Qbase = Qb + (size_t)(b * NH_ + h) * S_ * HD_;
    const u16* Kbase = Kb + (size_t)(b * NKV_ + kv) * S_ * HD_;
    const u16* Vbase = Vt + (size_t)(b * NKV_ + kv) * HD_ * S_;
    u16* Pw = Ps + wave * 1024;

    // preload Q fragments: 2 row-subtiles x 4 K-steps (HD=128)
    Frag16 qf[2][4];
#pragma unroll
    for (int m = 0; m < 2; ++m)
#pragma unroll
        for (int ks = 0; ks < 4; ++ks) {
            int row = waveQ + m * 16 + lh;
            int k0  = ks * 32 + (hi ? 8 : 0);
            qf[m][ks].u[0] = *(const uint4*)(Qbase + (size_t)row * HD_ + k0);
            qf[m][ks].u[1] = *(const uint4*)(Qbase + (size_t)row * HD_ + k0 + 16);
        }

    FragC o[2][8];
    float mrow[2][8], lrow[2][8];
#pragma unroll
    for (int m = 0; m < 2; ++m)
#pragma unroll
        for (int r = 0; r < 8; ++r) {
            mrow[m][r] = -1.0e30f;
            lrow[m][r] = 0.0f;
#pragma unroll
            for (int n = 0; n < 8; ++n) o[m][n].f[r] = 0.0f;
        }

    const float SCL = 0.08838834764831845f * 1.4426950408889634f; // rsqrt(HD)*log2e
    const int nkt = (qblk + 128) >> 5;              // causal: keys <= last q row

    if (wave == 0) {                                // prologue DMA: tile 0
        tdm_load_2d(lds_off_of(&Ks[0][0]),  Kbase, 128, 32, 128);
        tdm_load_2d(lds_off_of(&Vts[0][0]), Vbase, 32, 128, S_);
    }

    for (int kt = 0; kt < nkt; ++kt) {
        const int cur = kt & 1;
        if (wave == 0) {
            if (kt + 1 < nkt) {
                tdm_load_2d(lds_off_of(&Ks[cur ^ 1][0]),
                            Kbase + (size_t)((kt + 1) * 32) * HD_, 128, 32, 128);
                tdm_load_2d(lds_off_of(&Vts[cur ^ 1][0]),
                            Vbase + (kt + 1) * 32, 32, 128, S_);
                __builtin_amdgcn_s_wait_tensorcnt(2);
            } else {
                __builtin_amdgcn_s_wait_tensorcnt(0);
            }
        }
        __syncthreads();
        const u16* Ksb  = &Ks[cur][0];
        const u16* Vtsb = &Vts[cur][0];

        // ---- scores S = Q K^T : preload 8 B-frags, then 16 wmma ----
        Frag16 bf[4][2];
#pragma unroll
        for (int ks = 0; ks < 4; ++ks)
#pragma unroll
            for (int n = 0; n < 2; ++n) {
                int col = n * 16 + lh;
                int k0  = ks * 32 + (hi ? 8 : 0);
                bf[ks][n].u[0] = *(const uint4*)(Ksb + col * 128 + k0);
                bf[ks][n].u[1] = *(const uint4*)(Ksb + col * 128 + k0 + 16);
            }
        FragC sc[2][2];
#pragma unroll
        for (int m = 0; m < 2; ++m)
#pragma unroll
            for (int n = 0; n < 2; ++n)
#pragma unroll
                for (int r = 0; r < 8; ++r) sc[m][n].f[r] = 0.0f;
#pragma unroll
        for (int ks = 0; ks < 4; ++ks)
#pragma unroll
            for (int n = 0; n < 2; ++n)
#pragma unroll
                for (int m = 0; m < 2; ++m)
                    sc[m][n].v = __builtin_amdgcn_wmma_f32_16x16x32_bf16(
                        false, qf[m][ks].v, false, bf[ks][n].v, (short)0,
                        sc[m][n].v, false, false);

        // ---- mask + online softmax + write P scratch ----
        const int colb = kt * 32 + lh;
#pragma unroll
        for (int m = 0; m < 2; ++m) {
#pragma unroll
            for (int r = 0; r < 8; ++r) {
                int qr = waveQ + m * 16 + r + rofs;
                float x0 = sc[m][0].f[r] * SCL;
                float x1 = sc[m][1].f[r] * SCL;
                if (colb > qr)      x0 = -1.0e30f;
                if (colb + 16 > qr) x1 = -1.0e30f;
                float mx = fmaxf(x0, x1);
                mx = fmaxf(mx, __shfl_xor(mx, 1, 16));
                mx = fmaxf(mx, __shfl_xor(mx, 2, 16));
                mx = fmaxf(mx, __shfl_xor(mx, 4, 16));
                mx = fmaxf(mx, __shfl_xor(mx, 8, 16));
                float mold = mrow[m][r];
                float mnew = fmaxf(mold, mx);
                float p0 = exp2f(x0 - mnew);
                float p1 = exp2f(x1 - mnew);
                float rs = p0 + p1;
                rs += __shfl_xor(rs, 1, 16);
                rs += __shfl_xor(rs, 2, 16);
                rs += __shfl_xor(rs, 4, 16);
                rs += __shfl_xor(rs, 8, 16);
                float fcorr = exp2f(mold - mnew);
                mrow[m][r] = mnew;
                lrow[m][r] = lrow[m][r] * fcorr + rs;
#pragma unroll
                for (int n = 0; n < 8; ++n) o[m][n].f[r] *= fcorr;
                int prow = m * 16 + r + rofs;
                Pw[prow * 32 + lh]      = f2bf(p0);
                Pw[prow * 32 + 16 + lh] = f2bf(p1);
            }
        }
        __syncthreads();

        // ---- O += P V : preload pf/vf frags, then 16 wmma ----
        Frag16 pf[2], vf[8];
#pragma unroll
        for (int m = 0; m < 2; ++m) {
            int row = m * 16 + lh;
            int k0  = hi ? 8 : 0;
            pf[m].u[0] = *(const uint4*)(Pw + row * 32 + k0);
            pf[m].u[1] = *(const uint4*)(Pw + row * 32 + k0 + 16);
        }
#pragma unroll
        for (int n = 0; n < 8; ++n) {
            int dcol = n * 16 + lh;
            int kk   = hi ? 8 : 0;
            vf[n].u[0] = *(const uint4*)(Vtsb + dcol * 32 + kk);
            vf[n].u[1] = *(const uint4*)(Vtsb + dcol * 32 + kk + 16);
        }
#pragma unroll
        for (int n = 0; n < 8; ++n)
#pragma unroll
            for (int m = 0; m < 2; ++m)
                o[m][n].v = __builtin_amdgcn_wmma_f32_16x16x32_bf16(
                    false, pf[m].v, false, vf[n].v, (short)0, o[m][n].v,
                    false, false);
        __syncthreads();
    }

    // ---- normalize and store attn[B*S][NH*HD] in bf16 ----
#pragma unroll
    for (int m = 0; m < 2; ++m)
#pragma unroll
        for (int r = 0; r < 8; ++r) {
            float inv = 1.0f / lrow[m][r];
            int qr = waveQ + m * 16 + r + rofs;
            size_t grow = (size_t)(b * S_ + qr) * QKN_ + h * HD_;
#pragma unroll
            for (int n = 0; n < 8; ++n)
                attn[grow + n * 16 + lh] = f2bf(o[m][n].f[r] * inv);
        }
}

// ---------------------------------------------------------------------------
// Host launch
// ---------------------------------------------------------------------------
extern "C" void kernel_launch(void* const* d_in, const int* in_sizes, int n_in,
                              void* d_out, int out_size, void* d_ws,
                              size_t ws_size, hipStream_t stream) {
    (void)in_sizes; (void)n_in; (void)out_size; (void)ws_size;
    const float* hs = (const float*)d_in[0];
    // d_in[1] = attention_mask (causal; computed inline in flash_attn)
    const float* wq = (const float*)d_in[2];
    const float* wk = (const float*)d_in[3];
    const float* wv = (const float*)d_in[4];
    const float* wo = (const float*)d_in[5];

    char* ws = (char*)d_ws;
    size_t off = 0;
    auto alloc = [&](size_t bytes) -> void* {
        void* p = ws + off;
        off += (bytes + 255) & ~(size_t)255;
        return p;
    };
    u16* hsb   = (u16*)alloc((size_t)BS_ * H_ * 2);      // bf16 hidden
    u16* wqT   = (u16*)alloc((size_t)QKN_ * H_ * 2);     // [N][K]
    u16* wkT   = (u16*)alloc((size_t)KVN_ * H_ * 2);
    u16* wvT   = (u16*)alloc((size_t)KVN_ * H_ * 2);
    u16* woT   = (u16*)alloc((size_t)H_ * QKN_ * 2);
    u16* qproj = (u16*)alloc((size_t)BS_ * QKN_ * 2);
    u16* kproj = (u16*)alloc((size_t)BS_ * KVN_ * 2);
    u16* vproj = (u16*)alloc((size_t)BS_ * KVN_ * 2);
    u16* Qbuf  = (u16*)alloc((size_t)B_ * NH_ * S_ * HD_ * 2);
    u16* Kbuf  = (u16*)alloc((size_t)B_ * NKV_ * S_ * HD_ * 2);
    u16* Vtr   = (u16*)alloc((size_t)B_ * NKV_ * HD_ * S_ * 2);
    u16* attn  = (u16*)alloc((size_t)BS_ * QKN_ * 2);

    // 1) precision conversion (+ weight transpose)
    {
        int n = BS_ * H_;
        cvt_bf16<<<(n + 255) / 256, 256, 0, stream>>>(hs, hsb, n);
        int nq = H_ * QKN_;
        cvt_bf16_t<<<(nq + 255) / 256, 256, 0, stream>>>(wq, wqT, H_, QKN_);
        int nk = H_ * KVN_;
        cvt_bf16_t<<<(nk + 255) / 256, 256, 0, stream>>>(wk, wkT, H_, KVN_);
        cvt_bf16_t<<<(nk + 255) / 256, 256, 0, stream>>>(wv, wvT, H_, KVN_);
        int no = QKN_ * H_;
        cvt_bf16_t<<<(no + 255) / 256, 256, 0, stream>>>(wo, woT, QKN_, H_);
    }

    // 2) Q/K/V projections (bf16 WMMA GEMMs, TDM-staged)
    gemm_bf16_wmma<0><<<dim3(QKN_ / 128, BS_ / 128), 256, 0, stream>>>(
        hsb, wqT, qproj, BS_, QKN_, H_);
    gemm_bf16_wmma<0><<<dim3(KVN_ / 128, BS_ / 128), 256, 0, stream>>>(
        hsb, wkT, kproj, BS_, KVN_, H_);
    gemm_bf16_wmma<0><<<dim3(KVN_ / 128, BS_ / 128), 256, 0, stream>>>(
        hsb, wvT, vproj, BS_, KVN_, H_);

    // 3) RoPE + head-major reshape (K row-major, V transposed for WMMA B)
    rope_reshape<<<BS_, 128, 0, stream>>>(qproj, kproj, vproj, Qbuf, Kbuf, Vtr);

    // 4) causal GQA flash attention (TDM-staged K/V tiles)
    flash_attn<<<dim3(S_ / 128, NH_, B_), 128, 0, stream>>>(Qbuf, Kbuf, Vtr, attn);

    // 5) output projection, fp32 result
    gemm_bf16_wmma<1><<<dim3(H_ / 128, BS_ / 128), 256, 0, stream>>>(
        attn, woT, d_out, BS_, H_, QKN_);
}
